// BMManager_76476187673212
// MI455X (gfx1250) — compile-verified, matrix operand order checked by reading, not run on hardware
//
#include <hip/hip_runtime.h>
#include <hip/hip_bf16.h>
#include <math.h>

typedef __attribute__((ext_vector_type(16))) _Float16 v16h;
typedef __attribute__((ext_vector_type(8)))  _Float16 v8h;
typedef __attribute__((ext_vector_type(4)))  _Float16 v4h;
typedef __attribute__((ext_vector_type(8)))  float    v8f;

namespace {
constexpr int kB = 32;
constexpr int kS = 4096;
constexpr int kD = 512;
constexpr int kG = 128;
constexpr int kM = kB * kS;          // 131072 rows
constexpr int kLDW = 40;             // padded LDS row stride (halfs) for 32-wide k-slice
constexpr unsigned kKeep16 = 58982u; // 0.9 * 2^16 (pair-wise 16-bit thresholds)
constexpr float kScale = 1.0f / 0.9f;
}

// Full-strength hash for the (tiny) noise kernel.
__device__ __forceinline__ unsigned hash_u32(unsigned x) {
  x ^= x >> 17; x *= 0xed5ad4bbu;
  x ^= x >> 11; x *= 0xac4c1b51u;
  x ^= x >> 15; x *= 0x31848babu;
  x ^= x >> 14;
  return x;
}

// Cheap 2-mul mixer: one hash yields two 16-bit keep decisions (hot path).
__device__ __forceinline__ unsigned pair_hash(unsigned x) {
  x *= 0x9e3779b1u; x ^= x >> 16;
  x *= 0x85ebca6bu; x ^= x >> 13;
  return x;
}

// ---------------------------------------------------------------------------
// K0: zero the global accumulators (sum, sumsq)
// ---------------------------------------------------------------------------
__global__ void bm_init_kernel(float* __restrict__ accum) {
  if (threadIdx.x < 2) accum[threadIdx.x] = 0.0f;
}

// ---------------------------------------------------------------------------
// K1: h = dropout(x) @ W^T + b  via f16 WMMA (f32 accumulate), plus sum/sumsq
// grid = kM/128 blocks, 256 threads (8 waves). Wave w owns 16 M-rows x 128 N.
// ---------------------------------------------------------------------------
__global__ __launch_bounds__(256)
void bm_gemm_kernel(const float* __restrict__ x, const float* __restrict__ W,
                    const float* __restrict__ bias, float* __restrict__ hbuf,
                    float* __restrict__ accum) {
  __shared__ _Float16 Wl[kG * kLDW];   // 128 x (32 + pad) halfs = 10 KB

  const int tid  = threadIdx.x;
  const int lane = tid & 31;
  const int wave = tid >> 5;
  const int lrow = lane & 15;          // M row within 16x16 tile (A), N row (B), N column (C)
  const int kh   = lane >> 4;          // which K-half the lane carries
  const int m0   = blockIdx.x * 128 + wave * 16;

  v8f zero = {};
  v8f acc[8];
  #pragma unroll
  for (int n = 0; n < 8; ++n) acc[n] = zero;

  for (int t = 0; t < kD / 32; ++t) {
    __syncthreads();
    // Stage W[:, t*32 .. t*32+32) into LDS as f16 (256 threads x 4 slots of 4)
    #pragma unroll
    for (int i = 0; i < 4; ++i) {
      int slot = tid + 256 * i;            // 0..1023
      int row  = slot >> 3;
      int c    = (slot & 7) * 4;
      float4 wv = *(const float4*)(W + row * kD + t * 32 + c);
      v4h hv;
      hv[0] = (_Float16)wv.x; hv[1] = (_Float16)wv.y;
      hv[2] = (_Float16)wv.z; hv[3] = (_Float16)wv.w;
      *(v4h*)(&Wl[row * kLDW + c]) = hv;
    }
    __syncthreads();

    // A fragment 16x32 (f16) straight from global x, dropout applied.
    // ISA layout: lane holds K = [8*kh, 8*kh+8) and [16+8*kh, 16+8*kh+8).
    const int m     = m0 + lrow;
    const int kbase = t * 32 + kh * 8;
    const unsigned xoff = (unsigned)m * kD + kbase;
    float4 f0 = *(const float4*)(x + xoff);
    float4 f1 = *(const float4*)(x + xoff + 4);
    float4 f2 = *(const float4*)(x + xoff + 16);
    float4 f3 = *(const float4*)(x + xoff + 20);
    float r0[8] = {f0.x, f0.y, f0.z, f0.w, f1.x, f1.y, f1.z, f1.w};
    float r1[8] = {f2.x, f2.y, f2.z, f2.w, f3.x, f3.y, f3.z, f3.w};

    // One hash per element *pair*: low/high 16 bits give two keep decisions.
    const unsigned poff = xoff >> 1;        // xoff is a multiple of 8 -> even
    v16h a;
    #pragma unroll
    for (int j = 0; j < 4; ++j) {
      unsigned h0 = pair_hash(poff + j);        // elems xoff+2j, xoff+2j+1
      unsigned h1 = pair_hash(poff + 8 + j);    // elems xoff+16+2j, +1
      float e0 = ((h0 & 0xFFFFu) < kKeep16) ? r0[2 * j]     * kScale : 0.0f;
      float e1 = ((h0 >> 16)     < kKeep16) ? r0[2 * j + 1] * kScale : 0.0f;
      float e2 = ((h1 & 0xFFFFu) < kKeep16) ? r1[2 * j]     * kScale : 0.0f;
      float e3 = ((h1 >> 16)     < kKeep16) ? r1[2 * j + 1] * kScale : 0.0f;
      a[2 * j]         = (_Float16)e0;
      a[2 * j + 1]     = (_Float16)e1;
      a[8 + 2 * j]     = (_Float16)e2;
      a[8 + 2 * j + 1] = (_Float16)e3;
    }

    // 8 B fragments (W rows = N) from LDS, 2x ds_load_b128 each.
    #pragma unroll
    for (int n = 0; n < 8; ++n) {
      const _Float16* bp = &Wl[(n * 16 + lrow) * kLDW + kh * 8];
      v8h blo = *(const v8h*)bp;
      v8h bhi = *(const v8h*)(bp + 16);
      v16h bf;
      #pragma unroll
      for (int j = 0; j < 8; ++j) { bf[j] = blo[j]; bf[8 + j] = bhi[j]; }
      acc[n] = __builtin_amdgcn_wmma_f32_16x16x32_f16(
          false, a, false, bf, (short)0, acc[n], false, false);
    }
  }

  // Epilogue: bias add, store h, accumulate sum / sumsq.
  float bv[8];
  #pragma unroll
  for (int n = 0; n < 8; ++n) bv[n] = bias[n * 16 + lrow];

  float s1 = 0.0f, s2 = 0.0f;
  #pragma unroll
  for (int n = 0; n < 8; ++n) {
    #pragma unroll
    for (int r = 0; r < 8; ++r) {
      float v = acc[n][r] + bv[n];
      int mrow = m0 + kh * 8 + r;          // C layout: lanes 0-15 -> M 0..7, 16-31 -> 8..15
      hbuf[(size_t)mrow * kG + n * 16 + lrow] = v;
      s1 += v;
      s2 += v * v;
    }
  }
  #pragma unroll
  for (int off = 16; off > 0; off >>= 1) {
    s1 += __shfl_down(s1, off, 32);
    s2 += __shfl_down(s2, off, 32);
  }
  if (lane == 0) {
    atomicAdd(&accum[0], s1);
    atomicAdd(&accum[1], s2);
  }
}

// ---------------------------------------------------------------------------
// K2: noise[g] = 0.5*mean + std*z  (mean/std detached, Box-Muller gaussian)
// ---------------------------------------------------------------------------
__global__ void bm_noise_kernel(const float* __restrict__ accum,
                                float* __restrict__ noise) {
  int g = threadIdx.x;
  float sum = accum[0], sumsq = accum[1];
  const double N = (double)kM * (double)kG;
  double mean_all = (double)sum / N;
  double var = ((double)sumsq - (double)sum * (double)sum / N) / (N - 1.0);
  float stdv = sqrtf(fmaxf((float)var, 0.0f)) * (1.0f / 5.0f);
  float mn   = (float)mean_all * (1.0f / 10.0f);
  unsigned h1 = hash_u32(0xC0FFEEu + 2u * (unsigned)g);
  unsigned h2 = hash_u32(0xBADF00Du + 2u * (unsigned)g + 1u);
  float u1 = ((float)h1 + 0.5f) * (1.0f / 4294967296.0f);
  float u2 = ((float)h2 + 0.5f) * (1.0f / 4294967296.0f);
  float z = sqrtf(-2.0f * logf(u1)) * cosf(6.28318530718f * u2);
  noise[g] = 0.5f * mn + stdv * z;  // mean + std*z - 0.5*mean
}

// ---------------------------------------------------------------------------
// K3: last_start[b,s] = cummax over t<=s of (start(t) ? t : 0); one block per b
// ---------------------------------------------------------------------------
__global__ __launch_bounds__(256)
void bm_scan_kernel(const unsigned char* __restrict__ mask,
                    int* __restrict__ ls) {
  __shared__ int sh[256];
  __shared__ int carry_sh;
  const int b = blockIdx.x;
  const int tid = threadIdx.x;
  if (tid == 0) carry_sh = 0;
  __syncthreads();

  for (int c = 0; c < kS / 256; ++c) {
    int s = c * 256 + tid;
    int isStart = (s == 0) ? 1 : (mask[b * kS + s - 1] != 0);
    sh[tid] = isStart ? s : 0;
    __syncthreads();
    #pragma unroll
    for (int off = 1; off < 256; off <<= 1) {
      int t = (tid >= off) ? sh[tid - off] : 0;
      __syncthreads();
      if (t > sh[tid]) sh[tid] = t;
      __syncthreads();
    }
    int cprev = carry_sh;
    int val = sh[tid] > cprev ? sh[tid] : cprev;
    ls[b * kS + s] = val;
    __syncthreads();
    if (tid == 255) carry_sh = val;
    __syncthreads();
  }
}

// ---------------------------------------------------------------------------
// K4: out[b,s,:] = h[b, last_start[b,s], :] + noise[:]  (float4 vectorized)
// ---------------------------------------------------------------------------
__global__ __launch_bounds__(256)
void bm_gather_kernel(const float* __restrict__ hbuf,
                      const int* __restrict__ ls,
                      const float* __restrict__ noise,
                      float* __restrict__ out) {
  const int q   = blockIdx.x * 256 + threadIdx.x;  // float4 index
  const int row = q >> 5;                          // kG/4 = 32 float4 per row
  const int c4  = q & 31;
  const int b   = row >> 12;                       // S = 4096
  const int src = ls[row];
  const size_t so = ((size_t)(b << 12) + (size_t)src) * kG + (size_t)c4 * 4;
  float4 h4 = *(const float4*)(hbuf + so);
  float4 n4 = *(const float4*)(noise + c4 * 4);
  float4 o;
  o.x = h4.x + n4.x; o.y = h4.y + n4.y;
  o.z = h4.z + n4.z; o.w = h4.w + n4.w;
  *(float4*)(out + (size_t)q * 4) = o;
}

// ---------------------------------------------------------------------------
extern "C" void kernel_launch(void* const* d_in, const int* in_sizes, int n_in,
                              void* d_out, int out_size, void* d_ws, size_t ws_size,
                              hipStream_t stream) {
  const float*         x    = (const float*)d_in[0];
  const unsigned char* mask = (const unsigned char*)d_in[1];  // jax bool_ = 1 byte
  const float*         W    = (const float*)d_in[2];
  const float*         bias = (const float*)d_in[3];
  float* out = (float*)d_out;

  // Workspace layout
  char* ws = (char*)d_ws;
  float* hbuf  = (float*)ws;                                   // kM*kG floats (64 MB)
  int*   ls    = (int*)(ws + (size_t)kM * kG * sizeof(float)); // kM ints (512 KB)
  float* noise = (float*)((char*)ls + (size_t)kM * sizeof(int));
  float* accum = noise + 256;

  bm_init_kernel<<<1, 32, 0, stream>>>(accum);
  bm_gemm_kernel<<<kM / 128, 256, 0, stream>>>(x, W, bias, hbuf, accum);
  bm_noise_kernel<<<1, kG, 0, stream>>>(accum, noise);
  bm_scan_kernel<<<kB, 256, 0, stream>>>(mask, ls);
  bm_gather_kernel<<<(kM * (kG / 4)) / 256, 256, 0, stream>>>(hbuf, ls, noise, out);
}
